// RNModel_23545010717284
// MI455X (gfx1250) — matrix operand order repaired and checked by reading, not compiled, not run on hardware
//
#include <hip/hip_runtime.h>

// 1 = stage g_theta weights with the Tensor Data Mover (tensor_load_to_lds),
// 0 = fall back to per-lane global_load_async_to_lds_b128 (proven to compile).
#ifndef USE_TDM
#define USE_TDM 1
#endif

typedef __attribute__((ext_vector_type(16))) _Float16 v16h;
typedef __attribute__((ext_vector_type(8)))  float    v8f;
typedef __attribute__((ext_vector_type(4)))  unsigned int v4u;
typedef __attribute__((ext_vector_type(8)))  int      v8i;
typedef __attribute__((ext_vector_type(4)))  int      v4i;

namespace {

constexpr int kPad   = 264;   // LDS row stride in halves (256 + 8 -> conflict-free b128 loads)
constexpr int kPairs = 625;   // 25*25 relation pairs per batch element
constexpr int kMT    = 128;   // pair-rows per workgroup (8 waves x 16 rows)

union AF {
  v16h v;
  uint4 u[2];
};

// ---------------------------------------------------------------------------
// Conv 3x3 stride-2 pad-1 + bias + ReLU, with per-channel sum/sumsq stats.
// BN of the *previous* layer is folded into the input load via (scale,shift).
// One block handles a pixel chunk of a single (batch, out-channel) plane.
// ---------------------------------------------------------------------------
__global__ void conv_relu_stats(const float* __restrict__ in,
                                const float2* __restrict__ inbn,  // null for conv1
                                const float* __restrict__ w,      // [OC][IC][3][3]
                                const float* __restrict__ bias,
                                float* __restrict__ out,
                                float* __restrict__ stats,        // [24][2] atomics
                                int IC, int IH, int OH) {
  const int b  = blockIdx.z;
  const int oc = blockIdx.y;
  const int p  = blockIdx.x * blockDim.x + threadIdx.x;
  const int IW = IH, OW = OH;
  float val = 0.f;
  if (p < OH * OW) {
    const int oy = p / OW, ox = p - oy * OW;
    float acc = 0.f;
    for (int ic = 0; ic < IC; ++ic) {
      float sc = 1.f, sh = 0.f;
      if (inbn) { float2 t = inbn[ic]; sc = t.x; sh = t.y; }
      const float* ip = in + ((size_t)(b * IC + ic)) * IH * IW;
      const float* wp = w + (oc * IC + ic) * 9;
      for (int ky = 0; ky < 3; ++ky) {
        const int iy = oy * 2 + ky - 1;
        if (iy < 0 || iy >= IH) continue;
        for (int kx = 0; kx < 3; ++kx) {
          const int ix = ox * 2 + kx - 1;
          if (ix < 0 || ix >= IW) continue;
          acc += wp[ky * 3 + kx] * (ip[iy * IW + ix] * sc + sh);
        }
      }
    }
    acc += bias[oc];
    val = acc > 0.f ? acc : 0.f;
    out[((size_t)(b * 24 + oc)) * OH * OW + p] = val;
  }
  __shared__ float2 red[256];
  red[threadIdx.x] = make_float2(val, val * val);
  __syncthreads();
  for (int s = 128; s > 0; s >>= 1) {
    if ((int)threadIdx.x < s) {
      red[threadIdx.x].x += red[threadIdx.x + s].x;
      red[threadIdx.x].y += red[threadIdx.x + s].y;
    }
    __syncthreads();
  }
  if (threadIdx.x == 0) {
    atomicAdd(&stats[oc * 2 + 0], red[0].x);
    atomicAdd(&stats[oc * 2 + 1], red[0].y);
  }
}

__global__ void finalize_bn(const float* __restrict__ stats,
                            const float* __restrict__ g,
                            const float* __restrict__ be,
                            float2* __restrict__ bn, float inv_cnt) {
  const int c = threadIdx.x;
  if (c < 24) {
    const float mean = stats[c * 2 + 0] * inv_cnt;
    const float var  = stats[c * 2 + 1] * inv_cnt - mean * mean;
    const float rs   = rsqrtf(var + 1e-5f);
    const float sc   = g[c] * rs;
    bn[c] = make_float2(sc, be[c] - mean * sc);
  }
}

// ---------------------------------------------------------------------------
// Prep: x_flat[B][25][26] in f16 (BN4 applied, coord channels appended),
// ques f16, g-weights converted to f16 (layer-1 K padded 62 -> 64).
// ---------------------------------------------------------------------------
__global__ void build_xf(const float* __restrict__ x4, const float2* __restrict__ bn4,
                         _Float16* __restrict__ xf) {
  const int idx = blockIdx.x * 256 + threadIdx.x;
  if (idx >= 1024 * 25 * 26) return;
  const int c = idx % 26;
  const int t = idx / 26;
  const int pos = t % 25;
  const int b = t / 25;
  const int h = pos / 5, wcol = pos % 5;
  float v;
  if (c < 24) {
    const float2 s = bn4[c];
    v = x4[((size_t)(b * 24 + c)) * 25 + pos] * s.x + s.y;
  } else if (c == 24) {
    v = -2.5f + 1.25f * (float)wcol;   // cx: varies along width
  } else {
    v = -2.5f + 1.25f * (float)h;      // cy: varies along height
  }
  xf[idx] = (_Float16)v;
}

__global__ void cvt_f32_f16(const float* __restrict__ src, _Float16* __restrict__ dst, int n) {
  const int idx = blockIdx.x * 256 + threadIdx.x;
  if (idx < n) dst[idx] = (_Float16)src[idx];
}

__global__ void cvt_w1(const float* __restrict__ gW1, _Float16* __restrict__ w1h) {
  const int idx = blockIdx.x * 256 + threadIdx.x;   // 256 * 64
  if (idx >= 256 * 64) return;
  const int k = idx & 63, row = idx >> 6;
  w1h[idx] = (k < 62) ? (_Float16)gW1[row * 62 + k] : (_Float16)0.f;
}

// ---------------------------------------------------------------------------
// Weight staging: TDM descriptor moving `nrows` weight rows of KH halves into
// LDS, inserting 4 dwords of padding after each row (-> kPad-half row stride).
// ---------------------------------------------------------------------------
__device__ __forceinline__ void stage_issue(_Float16* __restrict__ dst,
                                            const _Float16* __restrict__ src,
                                            int KH, int nrows, int tid, int wave) {
#if USE_TDM
  if (wave == 0) {
    const unsigned lds = (unsigned)(unsigned long long)dst;
    const unsigned long long ga = (unsigned long long)src;
    v4u g0;
    g0[0] = 1u;                                        // count=1 (valid user D#)
    g0[1] = lds;                                       // lds_addr (bytes)
    g0[2] = (unsigned)ga;                              // global_addr[31:0]
    g0[3] = ((unsigned)(ga >> 32) & 0x01ffffffu)       // global_addr[56:32]
            | (2u << 30);                              // type = 2 ("image")
    const unsigned padInt = (KH == 64) ? 4u : 6u;      // row = 32 / 128 dwords
    const unsigned padAmt = 3u;                        // pad 4 dwords (8 halves)
    v8i g1;
    g1[0] = (int)((1u << 16)                           // data_size = 2 bytes
                  | (1u << 20)                         // pad_enable
                  | (padInt << 22) | (padAmt << 25));
    g1[1] = (int)(((unsigned)KH & 0xffffu) << 16);     // tensor_dim0[15:0]
    g1[2] = (int)((((unsigned)KH >> 16) & 0xffffu)     // tensor_dim0[31:16]
                  | (256u << 16));                     // tensor_dim1[15:0]
    g1[3] = (int)(((unsigned)KH & 0xffffu) << 16);     // tile_dim0
    g1[4] = (int)((unsigned)nrows & 0xffffu);          // tile_dim1 (tile_dim2=0)
    g1[5] = (int)(unsigned)KH;                         // tensor_dim0_stride[31:0]
    g1[6] = 0;
    g1[7] = 0;
    v4i z4 = {0, 0, 0, 0};
#if __clang_major__ >= 23
    v8i z8 = {0, 0, 0, 0, 0, 0, 0, 0};
    __builtin_amdgcn_tensor_load_to_lds(g0, g1, z4, z4, z8, 0);
#else
    __builtin_amdgcn_tensor_load_to_lds(g0, g1, z4, z4, 0);
#endif
  }
#else
  (void)wave;
  const int cpr = KH >> 3;            // 16-byte chunks per weight row
  const int total = nrows * cpr;
  for (int c = tid; c < total; c += 256) {
    const int row = c / cpr;
    const int off = (c - row * cpr) << 3;   // halves
    const unsigned ldsAddr = (unsigned)(unsigned long long)(dst + row * kPad + off);
    const int goff = (row * KH + off) * 2;  // byte offset from weight base
    asm volatile("global_load_async_to_lds_b128 %0, %1, %2 offset:0"
                 :: "v"(ldsAddr), "v"(goff), "s"(src)
                 : "memory");
  }
#endif
}

__device__ __forceinline__ void stage_wait(int wave) {
#if USE_TDM
  if (wave == 0) __builtin_amdgcn_s_wait_tensorcnt(0);
#else
  (void)wave;
  asm volatile("s_wait_asynccnt 0x0" ::: "memory");
#endif
}

// ---------------------------------------------------------------------------
// g_theta layer: each wave pushes its 16 rows through 8 N-tiles (one weight
// half), running TWO independent WMMA accumulator chains for ILP.
// ---------------------------------------------------------------------------
template <int KCH, bool LAST>
__device__ __forceinline__ void run_layer(const _Float16* __restrict__ actIn,
                                          const _Float16* __restrict__ wl,
                                          const float* __restrict__ bias,
                                          _Float16* __restrict__ actOut,
                                          int wave, int lane, int mt, int b,
                                          float* __restrict__ xg, int ntBase) {
  const int lg = lane >> 4;       // half-wave group (0/1)
  const int ll = lane & 15;
  AF a[KCH];                      // A fragments: 16 rows x (KCH*32) halves
  const _Float16* arow = actIn + (wave * 16 + ll) * kPad;
#pragma unroll
  for (int k = 0; k < KCH; ++k) {
    a[k].u[0] = *(const uint4*)(arow + k * 32 + 8 * lg);
    a[k].u[1] = *(const uint4*)(arow + k * 32 + 8 * lg + 16);
  }
  for (int np = 0; np < 4; ++np) {
    const int nt0 = ntBase + np * 2;
    const int nc0 = nt0 * 16 + ll;
    const int nc1 = nc0 + 16;
    const float bv0 = bias[nc0];
    const float bv1 = bias[nc1];
    const _Float16* wr0 = wl + nc0 * kPad;
    const _Float16* wr1 = wl + nc1 * kPad;
    v8f acc0 = {};
    v8f acc1 = {};
#pragma unroll
    for (int k = 0; k < KCH; ++k) {
      AF b0, b1;
      b0.u[0] = *(const uint4*)(wr0 + k * 32 + 8 * lg);
      b0.u[1] = *(const uint4*)(wr0 + k * 32 + 8 * lg + 16);
      b1.u[0] = *(const uint4*)(wr1 + k * 32 + 8 * lg);
      b1.u[1] = *(const uint4*)(wr1 + k * 32 + 8 * lg + 16);
      acc0 = __builtin_amdgcn_wmma_f32_16x16x32_f16(false, a[k].v, false, b0.v,
                                                    (short)0, acc0, false, false);
      acc1 = __builtin_amdgcn_wmma_f32_16x16x32_f16(false, a[k].v, false, b1.v,
                                                    (short)0, acc1, false, false);
    }
    if (!LAST) {
#pragma unroll
      for (int r = 0; r < 8; ++r) {
        float h0 = acc0[r] + bv0;
        float h1 = acc1[r] + bv1;
        h0 = h0 > 0.f ? h0 : 0.f;
        h1 = h1 > 0.f ? h1 : 0.f;
        const int mloc = wave * 16 + r + 8 * lg;
        actOut[mloc * kPad + nc0] = (_Float16)h0;
        actOut[mloc * kPad + nc1] = (_Float16)h1;
      }
    } else {
      float s0 = 0.f, s1 = 0.f;
#pragma unroll
      for (int r = 0; r < 8; ++r) {
        float h0 = acc0[r] + bv0;
        float h1 = acc1[r] + bv1;
        h0 = h0 > 0.f ? h0 : 0.f;
        h1 = h1 > 0.f ? h1 : 0.f;
        const int prow = mt * kMT + wave * 16 + r + 8 * lg;
        if (prow < kPairs) { s0 += h0; s1 += h1; }
      }
      s0 += __shfl_xor(s0, 16, 32);
      s1 += __shfl_xor(s1, 16, 32);
      if (lg == 0) {
        atomicAdd(&xg[b * 256 + nc0], s0);
        atomicAdd(&xg[b * 256 + nc1], s1);
      }
    }
  }
}

// ---------------------------------------------------------------------------
// g_theta: fused 4-layer MLP on 640K pair-rows.  Activations live entirely in
// LDS; weights are TDM-pipelined per 128-column half (one descriptor in
// flight while all waves run WMMA on the other half).
// ---------------------------------------------------------------------------
__global__ void __launch_bounds__(256) g_theta_kernel(
    const _Float16* __restrict__ xf,   // [B][25][26]
    const _Float16* __restrict__ qf,   // [B][10]
    const _Float16* __restrict__ w1h,  // [256][64]
    const _Float16* __restrict__ w2h,  // [256][256]
    const _Float16* __restrict__ w3h,
    const _Float16* __restrict__ w4h,
    const float* __restrict__ gb1, const float* __restrict__ gb2,
    const float* __restrict__ gb3, const float* __restrict__ gb4,
    float* __restrict__ xg) {
  extern __shared__ char smem[];
  _Float16* actA = (_Float16*)smem;                  // 128 x 264 halves
  _Float16* actB = actA + kMT * kPad;                // 128 x 264 halves
  _Float16* wl   = actB + kMT * kPad;                // 256 x 264 halves
  _Float16* wlHi = wl + 128 * kPad;                  // second 128-column half

  const int tid  = threadIdx.x;
  const int wave = tid >> 5;
  const int lane = tid & 31;
  const int b  = blockIdx.x;
  const int mt = blockIdx.y;
  const int rowbase = mt * kMT;

  // Kick off layer-1 weights immediately (K=64, whole 256-row tile).
  stage_issue(wl, w1h, 64, 256, tid, wave);

  // Build layer-1 A tile: row = pair (i,j) -> [x_flat[b,j] | x_flat[b,i] | ques[b] | 0]
  for (int idx = tid; idx < kMT * 64; idx += 256) {
    const int r = idx >> 6, c = idx & 63;
    const int prow = rowbase + r;
    _Float16 v = (_Float16)0.f;
    if (prow < kPairs) {
      const int i = prow / 25, j = prow - i * 25;
      if (c < 26)      v = xf[(b * 25 + j) * 26 + c];
      else if (c < 52) v = xf[(b * 25 + i) * 26 + (c - 26)];
      else if (c < 62) v = qf[b * 10 + (c - 52)];
    }
    actA[r * kPad + c] = v;
  }
  stage_wait(wave);
  __syncthreads();

  // ---- L1 (K=64): both halves of W1 already resident ----
  run_layer<2, false>(actA, wl, gb1, actB, wave, lane, mt, b, nullptr, 0);
  run_layer<2, false>(actA, wl, gb1, actB, wave, lane, mt, b, nullptr, 8);
  __syncthreads();

  // ---- L2..L4: half-buffer TDM pipeline, one descriptor in flight ----
  stage_issue(wl, w2h, 256, 128, tid, wave);           // W2 half0
  stage_wait(wave);
  stage_issue(wlHi, w2h + 128 * 256, 256, 128, tid, wave);  // W2 half1 in flight
  __syncthreads();
  run_layer<8, false>(actB, wl, gb2, actA, wave, lane, mt, b, nullptr, 0);
  __syncthreads();
  stage_wait(wave);
  stage_issue(wl, w3h, 256, 128, tid, wave);           // W3 half0 in flight
  __syncthreads();
  run_layer<8, false>(actB, wl, gb2, actA, wave, lane, mt, b, nullptr, 8);
  __syncthreads();
  stage_wait(wave);
  stage_issue(wlHi, w3h + 128 * 256, 256, 128, tid, wave);  // W3 half1 in flight
  __syncthreads();
  run_layer<8, false>(actA, wl, gb3, actB, wave, lane, mt, b, nullptr, 0);
  __syncthreads();
  stage_wait(wave);
  stage_issue(wl, w4h, 256, 128, tid, wave);           // W4 half0 in flight
  __syncthreads();
  run_layer<8, false>(actA, wl, gb3, actB, wave, lane, mt, b, nullptr, 8);
  __syncthreads();
  stage_wait(wave);
  stage_issue(wlHi, w4h + 128 * 256, 256, 128, tid, wave);  // W4 half1 in flight
  __syncthreads();
  run_layer<8, true>(actB, wl, gb4, nullptr, wave, lane, mt, b, xg, 0);
  __syncthreads();
  stage_wait(wave);
  __syncthreads();
  run_layer<8, true>(actB, wl, gb4, nullptr, wave, lane, mt, b, xg, 8);
}

// ---------------------------------------------------------------------------
// f_phi: relu(L256) -> relu(L256) -> L10 -> log_softmax.  One block per row.
// ---------------------------------------------------------------------------
__global__ void __launch_bounds__(256) f_phi_kernel(
    const float* __restrict__ xg,
    const float* __restrict__ fW1, const float* __restrict__ fb1,
    const float* __restrict__ fW2, const float* __restrict__ fb2,
    const float* __restrict__ foW, const float* __restrict__ fob,
    float* __restrict__ out) {
  __shared__ float xrow[256];
  __shared__ float h1[256];
  __shared__ float logits[16];
  __shared__ float lse;
  const int b = blockIdx.x, t = threadIdx.x;
  xrow[t] = xg[b * 256 + t];
  __syncthreads();
  float acc = 0.f;
  const float* wr = fW1 + t * 256;
  for (int k = 0; k < 256; ++k) acc += wr[k] * xrow[k];
  acc += fb1[t];
  h1[t] = acc > 0.f ? acc : 0.f;
  __syncthreads();
  acc = 0.f;
  wr = fW2 + t * 256;
  for (int k = 0; k < 256; ++k) acc += wr[k] * h1[k];
  acc += fb2[t];
  __syncthreads();
  xrow[t] = acc > 0.f ? acc : 0.f;   // h2
  __syncthreads();
  if (t < 10) {
    acc = 0.f;
    wr = foW + t * 256;
    for (int k = 0; k < 256; ++k) acc += wr[k] * xrow[k];
    logits[t] = acc + fob[t];
  }
  __syncthreads();
  if (t == 0) {
    float m = logits[0];
    for (int i = 1; i < 10; ++i) m = fmaxf(m, logits[i]);
    float s = 0.f;
    for (int i = 0; i < 10; ++i) s += expf(logits[i] - m);
    lse = m + logf(s);
  }
  __syncthreads();
  if (t < 10) out[b * 10 + t] = logits[t] - lse;
}

}  // namespace

// ---------------------------------------------------------------------------
extern "C" void kernel_launch(void* const* d_in, const int* in_sizes, int n_in,
                              void* d_out, int out_size, void* d_ws, size_t ws_size,
                              hipStream_t stream) {
  (void)in_sizes; (void)n_in; (void)out_size; (void)ws_size;
  const float* img  = (const float*)d_in[0];
  const float* ques = (const float*)d_in[1];
  const float* cw1 = (const float*)d_in[2];  const float* cb1 = (const float*)d_in[3];
  const float* cw2 = (const float*)d_in[4];  const float* cb2 = (const float*)d_in[5];
  const float* cw3 = (const float*)d_in[6];  const float* cb3 = (const float*)d_in[7];
  const float* cw4 = (const float*)d_in[8];  const float* cb4 = (const float*)d_in[9];
  const float* bng1 = (const float*)d_in[10]; const float* bnb1 = (const float*)d_in[11];
  const float* bng2 = (const float*)d_in[12]; const float* bnb2 = (const float*)d_in[13];
  const float* bng3 = (const float*)d_in[14]; const float* bnb3 = (const float*)d_in[15];
  const float* bng4 = (const float*)d_in[16]; const float* bnb4 = (const float*)d_in[17];
  const float* gW1 = (const float*)d_in[18]; const float* gb1 = (const float*)d_in[19];
  const float* gW2 = (const float*)d_in[20]; const float* gb2 = (const float*)d_in[21];
  const float* gW3 = (const float*)d_in[22]; const float* gb3 = (const float*)d_in[23];
  const float* gW4 = (const float*)d_in[24]; const float* gb4 = (const float*)d_in[25];
  const float* fW1 = (const float*)d_in[26]; const float* fb1 = (const float*)d_in[27];
  const float* fW2 = (const float*)d_in[28]; const float* fb2 = (const float*)d_in[29];
  const float* foW = (const float*)d_in[30]; const float* fob = (const float*)d_in[31];

  // ---- workspace partition (deterministic) ----
  char* ws = (char*)d_ws;
  auto take = [&](size_t bytes) -> char* {
    char* p = ws;
    ws += (bytes + 255) & ~(size_t)255;
    return p;
  };
  float*    x1    = (float*)take((size_t)1024 * 24 * 38 * 38 * 4);
  float*    x2    = (float*)take((size_t)1024 * 24 * 19 * 19 * 4);
  float*    x3    = (float*)take((size_t)1024 * 24 * 10 * 10 * 4);
  float*    x4    = (float*)take((size_t)1024 * 24 * 5 * 5 * 4);
  float*    stats = (float*)take(4 * 48 * sizeof(float));       // 4 layers x 24ch x {sum,sumsq}
  float2*   bn    = (float2*)take(4 * 24 * sizeof(float2));
  _Float16* xf    = (_Float16*)take((size_t)1024 * 25 * 26 * 2);
  _Float16* qf    = (_Float16*)take((size_t)1024 * 10 * 2);
  _Float16* w1h   = (_Float16*)take((size_t)256 * 64 * 2);
  _Float16* w2h   = (_Float16*)take((size_t)256 * 256 * 2);
  _Float16* w3h   = (_Float16*)take((size_t)256 * 256 * 2);
  _Float16* w4h   = (_Float16*)take((size_t)256 * 256 * 2);
  float*    xg    = (float*)take((size_t)1024 * 256 * 4);

  hipMemsetAsync(stats, 0, 4 * 48 * sizeof(float), stream);
  hipMemsetAsync(xg, 0, (size_t)1024 * 256 * 4, stream);

  // ---- conv stack with fused BN-of-previous-layer ----
  {
    dim3 g((38 * 38 + 255) / 256, 24, 1024);
    conv_relu_stats<<<g, 256, 0, stream>>>(img, nullptr, cw1, cb1, x1, stats + 0, 3, 75, 38);
    finalize_bn<<<1, 32, 0, stream>>>(stats + 0, bng1, bnb1, bn + 0, 1.f / (1024.f * 38 * 38));
  }
  {
    dim3 g((19 * 19 + 255) / 256, 24, 1024);
    conv_relu_stats<<<g, 256, 0, stream>>>(x1, bn + 0, cw2, cb2, x2, stats + 48, 24, 38, 19);
    finalize_bn<<<1, 32, 0, stream>>>(stats + 48, bng2, bnb2, bn + 24, 1.f / (1024.f * 19 * 19));
  }
  {
    dim3 g(1, 24, 1024);
    conv_relu_stats<<<g, 256, 0, stream>>>(x2, bn + 24, cw3, cb3, x3, stats + 96, 24, 19, 10);
    finalize_bn<<<1, 32, 0, stream>>>(stats + 96, bng3, bnb3, bn + 48, 1.f / (1024.f * 10 * 10));
  }
  {
    dim3 g(1, 24, 1024);
    conv_relu_stats<<<g, 256, 0, stream>>>(x3, bn + 48, cw4, cb4, x4, stats + 144, 24, 10, 5);
    finalize_bn<<<1, 32, 0, stream>>>(stats + 144, bng4, bnb4, bn + 72, 1.f / (1024.f * 5 * 5));
  }

  // ---- f16 prep for WMMA stage ----
  build_xf<<<(1024 * 25 * 26 + 255) / 256, 256, 0, stream>>>(x4, bn + 72, xf);
  cvt_f32_f16<<<(1024 * 10 + 255) / 256, 256, 0, stream>>>(ques, qf, 1024 * 10);
  cvt_w1<<<(256 * 64 + 255) / 256, 256, 0, stream>>>(gW1, w1h);
  cvt_f32_f16<<<256, 256, 0, stream>>>(gW2, w2h, 256 * 256);
  cvt_f32_f16<<<256, 256, 0, stream>>>(gW3, w3h, 256 * 256);
  cvt_f32_f16<<<256, 256, 0, stream>>>(gW4, w4h, 256 * 256);

  // ---- g_theta: fused 4-layer WMMA MLP + pair-sum ----
  {
    const size_t ldsBytes = (size_t)(2 * kMT * kPad + 256 * kPad) * sizeof(_Float16); // 270336
    dim3 g(1024, (kPairs + kMT - 1) / kMT);   // 1024 x 5
    g_theta_kernel<<<g, 256, ldsBytes, stream>>>(xf, qf, w1h, w2h, w3h, w4h,
                                                 gb1, gb2, gb3, gb4, xg);
  }

  // ---- f_phi + log_softmax ----
  f_phi_kernel<<<1024, 256, 0, stream>>>(xg, fW1, fb1, fW2, fb2, foW, fob, (float*)d_out);
}
